// BioZorro_38062000177508
// MI455X (gfx1250) — matrix operand order repaired and checked by reading, not compiled, not run on hardware
//
#include <hip/hip_runtime.h>
#include <hip/hip_bf16.h>

typedef __attribute__((ext_vector_type(16))) _Float16 v16h;
typedef __attribute__((ext_vector_type(8)))  _Float16 h8v;
typedef __attribute__((ext_vector_type(8)))  float    v8f;
typedef __attribute__((ext_vector_type(2)))  _Float16 h2v;
typedef __attribute__((ext_vector_type(4)))  int      i4v;
typedef __attribute__((address_space(1)))    i4v      as1_i4v;
typedef __attribute__((address_space(3)))    i4v      as3_i4v;

#define WMMA_F16(a, b, c) \
  __builtin_amdgcn_wmma_f32_16x16x32_f16(false, (a), false, (b), (short)0, (c), false, false)

#if defined(__gfx1250__) && __has_builtin(__builtin_amdgcn_global_load_async_to_lds_b128) && __has_builtin(__builtin_amdgcn_s_wait_asynccnt)
#define HAS_ASYNC 1
#else
#define HAS_ASYNC 0
#endif

// ---- model dims ----
#define ZB     2
#define ZNR    768
#define ZNF    16
#define ZNA    768
#define ZNTOK  1552          // 768 + 16 + 768 (= 97 * 16)
#define ZDIM   512
#define ZRIN   1024
#define ZH     8
#define ZDH    64
#define ZIFF   1365
#define ZIFFP  1368          // padded to multiple of 8 halves (16B rows)
#define ZDEPTH 4
#define ZKB    97            // key blocks of 16
#define ZSCOL  1568          // 1552 padded to multiple of 32
#define ZCH    49            // K chunks of 32 covering 1568

// 16-byte global->LDS copy: async DMA on gfx1250, vector copy fallback.
__device__ __forceinline__ void cp16(const _Float16* g, _Float16* l) {
#if HAS_ASYNC
  __builtin_amdgcn_global_load_async_to_lds_b128(
      (as1_i4v*)(unsigned long long)(uintptr_t)g,
      (as3_i4v*)(unsigned)(uintptr_t)l,
      0, 0);
#else
  *(h8v*)l = *(const h8v*)g;
#endif
}
__device__ __forceinline__ void wait_async() {
#if HAS_ASYNC
  __builtin_amdgcn_s_wait_asynccnt(0);
#endif
}

// ---------------------------------------------------------------------------
// LayerNorm; optional f32 and/or f16 outputs, scatter mapping for output row.
// out_row = (row / rows_per_b) * out_bstride + out_row_off + row % rows_per_b
// ---------------------------------------------------------------------------
__global__ __launch_bounds__(256) void ln_kernel(
    const float* __restrict__ in, float* __restrict__ out32, _Float16* __restrict__ out16,
    const float* __restrict__ g, const float* __restrict__ bvec,
    int rows, int cols, int rows_per_b, long out_bstride, int out_row_off)
{
  int row = blockIdx.x;
  if (row >= rows) return;
  const float* x = in + (size_t)row * cols;
  __shared__ float red[8];
  __shared__ float stat[2];
  int tid = threadIdx.x;

  float s = 0.f;
  for (int c = tid; c < cols; c += 256) s += x[c];
#pragma unroll
  for (int off = 16; off > 0; off >>= 1) s += __shfl_xor(s, off, 32);
  if ((tid & 31) == 0) red[tid >> 5] = s;
  __syncthreads();
  if (tid == 0) { float t = 0.f; for (int i = 0; i < 8; ++i) t += red[i]; stat[0] = t / cols; }
  __syncthreads();
  float mu = stat[0];

  float vv = 0.f;
  for (int c = tid; c < cols; c += 256) { float d = x[c] - mu; vv += d * d; }
#pragma unroll
  for (int off = 16; off > 0; off >>= 1) vv += __shfl_xor(vv, off, 32);
  __syncthreads();
  if ((tid & 31) == 0) red[tid >> 5] = vv;
  __syncthreads();
  if (tid == 0) { float t = 0.f; for (int i = 0; i < 8; ++i) t += red[i]; stat[1] = t / cols; }
  __syncthreads();
  float rstd = rsqrtf(stat[1] + 1e-5f);

  size_t orow = (size_t)(row / rows_per_b) * (size_t)out_bstride + out_row_off + (row % rows_per_b);
  for (int c = tid; c < cols; c += 256) {
    float t = (x[c] - mu) * rstd * g[c];
    if (bvec) t += bvec[c];
    if (out32) out32[orow * cols + c] = t;
    if (out16) out16[orow * cols + c] = (_Float16)t;
  }
}

// ---------------------------------------------------------------------------
// WMMA GEMM: C = alpha * A(f16)[M,K] @ B + bias + resid.
// B is pre-transposed f16 [N,K] (row n holds K contiguous values), stride ldb.
// Outputs: optional f32 C and/or f16 Ch (both row stride N).
// 256 threads = 8 waves; 128x64 block tile; 32x32 wave tile (4 WMMA accs:
// 2 A-frags x 2 B-frags -> 8 ds_load_b128 per 4 WMMAs).
// Full tiles staged via async global->LDS b128 copies, double-buffered.
// ---------------------------------------------------------------------------
#define GBM 128
#define GBN 64
#define GBK 32
#define GKP 40   // LDS K stride (halves): 80B rows -> 16B aligned, 20-bank rotation

__global__ __launch_bounds__(256) void gemm_kernel(
    const _Float16* __restrict__ Ag, long lda,
    const _Float16* __restrict__ Bg, long ldb,
    const float* __restrict__ bias, const float* __restrict__ resid,
    float* __restrict__ C, _Float16* __restrict__ Ch,
    int M, int N, int K, float alpha)
{
  __shared__ _Float16 As[2][GBM * GKP];
  __shared__ _Float16 Bs[2][GBN * GKP];
  int tid  = threadIdx.x;
  int lane = tid & 31;
  int wave = tid >> 5;
  int hi   = lane >> 4;
  int ln15 = lane & 15;
  int bm0 = blockIdx.y * GBM;
  int bn0 = blockIdx.x * GBN;
  int wm  = (wave & 3) * 32;    // wave row offset (4 waves x 32 rows)
  int wn  = (wave >> 2) * 32;   // wave col offset (2 waves x 32 cols)

  int ar = tid >> 1;            // 0..127: A row; each thread copies 32B of a row
  int ak = (tid & 1) * 16;      // 16-half segment within the 32-half row
  int br = tid >> 2;            // 0..63:  B row (= output col n)
  int bk = (tid & 3) * 8;       // 8-half segment
  bool edgeMN = (bm0 + GBM > M) || (bn0 + GBN > N);

  v8f acc00 = {}, acc01 = {}, acc10 = {}, acc11 = {};
  int nk = (K + GBK - 1) / GBK;

  auto stage = [&](int kc, int buf) {
    int k0g = kc * GBK;
    if (!edgeMN && (k0g + GBK <= K)) {
      const _Float16* ga = Ag + (size_t)(bm0 + ar) * lda + k0g + ak;
      cp16(ga,     &As[buf][ar * GKP + ak]);
      cp16(ga + 8, &As[buf][ar * GKP + ak + 8]);
      cp16(Bg + (size_t)(bn0 + br) * ldb + k0g + bk, &Bs[buf][br * GKP + bk]);
    } else {
      int gr  = bm0 + ar;
      int grc = gr < M ? gr : M - 1;
      int gn  = bn0 + br;
      int gnc = gn < N ? gn : N - 1;
#pragma unroll
      for (int i = 0; i < 16; ++i) {
        int gk  = k0g + ak + i;
        int gkc = gk < K ? gk : K - 1;
        _Float16 va = Ag[(size_t)grc * lda + gkc];
        As[buf][ar * GKP + ak + i] = (gr < M && gk < K) ? va : (_Float16)0;
      }
#pragma unroll
      for (int i = 0; i < 8; ++i) {
        int gk  = k0g + bk + i;
        int gkc = gk < K ? gk : K - 1;
        _Float16 vb = Bg[(size_t)gnc * ldb + gkc];
        Bs[buf][br * GKP + bk + i] = (gn < N && gk < K) ? vb : (_Float16)0;
      }
    }
  };

  stage(0, 0);
  for (int kc = 0; kc < nk; ++kc) {
    int cur = kc & 1;
    wait_async();
    __syncthreads();
    if (kc + 1 < nk) stage(kc + 1, cur ^ 1);

    v16h a0, a1, b0, b1;
#pragma unroll
    for (int v = 0; v < 8; ++v) {
      int ka = ((v & 3) * 2) + hi * 8 + (v >> 2) * 16;      // A frag K map (ISA 7.12.2)
      h2v pa0 = *(const h2v*)&As[cur][(wm + ln15) * GKP + ka];
      a0[2 * v] = pa0[0]; a0[2 * v + 1] = pa0[1];
      h2v pa1 = *(const h2v*)&As[cur][(wm + 16 + ln15) * GKP + ka];
      a1[2 * v] = pa1[0]; a1[2 * v + 1] = pa1[1];
      int kb = ((v & 3) * 2) + hi * 8 + (v >> 2) * 16;      // B frag K map
      h2v pb0 = *(const h2v*)&Bs[cur][(wn + ln15) * GKP + kb];
      b0[2 * v] = pb0[0]; b0[2 * v + 1] = pb0[1];
      h2v pb1 = *(const h2v*)&Bs[cur][(wn + 16 + ln15) * GKP + kb];
      b1[2 * v] = pb1[0]; b1[2 * v + 1] = pb1[1];
    }
    acc00 = WMMA_F16(a0, b0, acc00);
    acc01 = WMMA_F16(a0, b1, acc01);
    acc10 = WMMA_F16(a1, b0, acc10);
    acc11 = WMMA_F16(a1, b1, acc11);
  }

  auto stc = [&](float av, int gr, int gc) {
    if (gr < M && gc < N) {
      float t = av * alpha;
      if (bias)  t += bias[gc];
      if (resid) t += resid[(size_t)gr * N + gc];
      if (C)  C[(size_t)gr * N + gc] = t;
      if (Ch) Ch[(size_t)gr * N + gc] = (_Float16)t;
    }
  };
#pragma unroll
  for (int v = 0; v < 8; ++v) {
    int r0 = bm0 + wm + v + hi * 8;                         // C map: M = v + 8*hi
    int r1 = r0 + 16;
    int c0 = bn0 + wn + ln15;
    int c1 = c0 + 16;
    stc(acc00[v], r0, c0);
    stc(acc01[v], r0, c1);
    stc(acc10[v], r1, c0);
    stc(acc11[v], r1, c1);
  }
}

// ---------------------------------------------------------------------------
// Fused masked attention for one (batch, head, 16-query block). f16 q/kv in,
// f16 out. Q pre-scaled. Scores/probs kept in LDS f16; softmax math in f32.
// mask_mode 0: zorro mask by token type; 1: pool mask (query row = ret type).
// ---------------------------------------------------------------------------
__device__ __forceinline__ int tok_type(int i) { return i < ZNR ? 0 : (i < ZNR + ZNF ? 2 : 1); }

__global__ __launch_bounds__(256) void attn_kernel(
    const _Float16* __restrict__ Q, long q_bstride, int Mq,
    const _Float16* __restrict__ KV, long kv_bstride,
    _Float16* __restrict__ O, long o_bstride, int mask_mode)
{
  __shared__ _Float16 S[16 * ZSCOL];     // scores -> probabilities (f16)
  __shared__ float Opart[4 * 16 * 16];   // K-half partial reduction

  int qb = blockIdx.x, h = blockIdx.y, b = blockIdx.z;
  int tid = threadIdx.x, lane = tid & 31, wave = tid >> 5;
  int hi = lane >> 4, ln15 = lane & 15;

  const _Float16* q    = Q  + (size_t)b * q_bstride;
  const _Float16* kmat = KV + (size_t)b * kv_bstride + h * ZDH;          // (j,d)
  const _Float16* vmat = KV + (size_t)b * kv_bstride + ZDIM + h * ZDH;   // (j,d)

  // ---- A fragments: q block 16x64 = two 16x32 chunks ----
  v16h aq[2];
  int qrow = qb * 16 + ln15;
  int qrc  = qrow < Mq ? qrow : Mq - 1;
  const _Float16* qrp = q + (size_t)qrc * ZDIM + h * ZDH;
#pragma unroll
  for (int dc = 0; dc < 2; ++dc) {
#pragma unroll
    for (int v = 0; v < 8; ++v) {
      int ka = ((v & 3) * 2) + hi * 8 + (v >> 2) * 16 + dc * 32;
      h2v p = *(const h2v*)&qrp[ka];
      bool in = (qrow < Mq);
      aq[dc][2 * v]     = in ? p[0] : (_Float16)0;
      aq[dc][2 * v + 1] = in ? p[1] : (_Float16)0;
    }
  }

  // ---- scores: each wave owns key blocks wave, wave+8, ... ----
  for (int kb = wave; kb < ZKB; kb += 8) {
    v8f acc = {};
#pragma unroll
    for (int dc = 0; dc < 2; ++dc) {
      v16h bf;
      int key = kb * 16 + ln15;
#pragma unroll
      for (int v = 0; v < 8; ++v) {
        int d = ((v & 3) * 2) + hi * 8 + (v >> 2) * 16 + dc * 32;
        h2v p = *(const h2v*)&kmat[(size_t)key * (2 * ZDIM) + d];
        bf[2 * v] = p[0]; bf[2 * v + 1] = p[1];
      }
      acc = WMMA_F16(aq[dc], bf, acc);
    }
    int j  = kb * 16 + ln15;
    int tj = tok_type(j);
#pragma unroll
    for (int v = 0; v < 8; ++v) {
      int r = v + hi * 8;
      bool ok;
      if (mask_mode == 0) {
        int ti = tok_type(qb * 16 + r);
        ok = (((ti == tj ? 1 : 0) | ti) == 2);      // replicates (same | ti) == 2
      } else {
        int rt = r;                                  // return-token type = row index
        ok = (((rt == tj ? 1 : 0) | rt) == 3);      // replicates (pm | rt) == 3
      }
      S[r * ZSCOL + j] = (_Float16)(ok ? acc[v] : -6.0e4f);
    }
  }
  __syncthreads();

  // ---- softmax: 16 threads per row, shfl width-16 reductions ----
  {
    int r  = tid >> 4;
    int c0 = tid & 15;
    float m = -3.4e38f;
    for (int j = c0; j < ZNTOK; j += 16) m = fmaxf(m, (float)S[r * ZSCOL + j]);
#pragma unroll
    for (int off = 8; off > 0; off >>= 1) m = fmaxf(m, __shfl_xor(m, off, 16));
    float sum = 0.f;
    for (int j = c0; j < ZNTOK; j += 16) sum += expf((float)S[r * ZSCOL + j] - m);
#pragma unroll
    for (int off = 8; off > 0; off >>= 1) sum += __shfl_xor(sum, off, 16);
    float inv = 1.f / sum;
    for (int j = c0; j < ZNTOK; j += 16)
      S[r * ZSCOL + j] = (_Float16)(expf((float)S[r * ZSCOL + j] - m) * inv);
    for (int j = ZNTOK + c0; j < ZSCOL; j += 16) S[r * ZSCOL + j] = (_Float16)0.f;
  }
  __syncthreads();

  // ---- O = P @ V: waves split 4 N-tiles x 2 K-halves ----
  int ntile = wave & 3;
  int khalf = wave >> 2;
  v8f oacc = {};
  int cbeg = khalf ? 25 : 0;
  int cend = khalf ? ZCH : 25;
  for (int ch = cbeg; ch < cend; ++ch) {
    v16h pf, vf;
#pragma unroll
    for (int v = 0; v < 8; ++v) {
      int ka = ((v & 3) * 2) + hi * 8 + (v >> 2) * 16;
      h2v pp = *(const h2v*)&S[ln15 * ZSCOL + ch * 32 + ka];
      pf[2 * v] = pp[0]; pf[2 * v + 1] = pp[1];
      int key  = ch * 32 + ((v & 3) * 2) + hi * 8 + (v >> 2) * 16;
      int k0c  = key     < ZNTOK ? key     : ZNTOK - 1;
      int k1c  = key + 1 < ZNTOK ? key + 1 : ZNTOK - 1;
      int d    = ntile * 16 + ln15;
      _Float16 v0 = vmat[(size_t)k0c * (2 * ZDIM) + d];
      _Float16 v1 = vmat[(size_t)k1c * (2 * ZDIM) + d];
      vf[2 * v]     = (key     < ZNTOK) ? v0 : (_Float16)0;
      vf[2 * v + 1] = (key + 1 < ZNTOK) ? v1 : (_Float16)0;
    }
    oacc = WMMA_F16(pf, vf, oacc);
  }
  if (khalf == 1) {
#pragma unroll
    for (int v = 0; v < 8; ++v)
      Opart[ntile * 256 + (v + hi * 8) * 16 + ln15] = oacc[v];
  }
  __syncthreads();
  if (khalf == 0) {
    _Float16* ob = O + (size_t)b * o_bstride;
#pragma unroll
    for (int v = 0; v < 8; ++v) {
      int rr = v + hi * 8;
      int qi = qb * 16 + rr;
      if (qi < Mq)
        ob[(size_t)qi * ZDIM + h * ZDH + ntile * 16 + ln15] =
            (_Float16)(oacc[v] + Opart[ntile * 256 + rr * 16 + ln15]);
    }
  }
}

// ---------------------------------------------------------------------------
// GEGLU: g16[r,c] = gelu_exact(h[r, IFF+c]) * h[r, c]    (f16 out, padded row)
// ---------------------------------------------------------------------------
__global__ __launch_bounds__(256) void geglu_kernel(
    const float* __restrict__ hbuf, _Float16* __restrict__ g16, long rows)
{
  long idx = (long)blockIdx.x * blockDim.x + threadIdx.x;
  long total = rows * ZIFF;
  if (idx >= total) return;
  long r = idx / ZIFF;
  int  c = (int)(idx - r * ZIFF);
  float x1 = hbuf[r * (2 * ZIFF) + c];
  float gt = hbuf[r * (2 * ZIFF) + ZIFF + c];
  float ge = 0.5f * gt * (1.f + erff(gt * 0.70710678118654752f));
  g16[r * ZIFFP + c] = (_Float16)(ge * x1);
}

// Weight convert + transpose: wT[n*ldt + k] = (f16) w[k*N + n]
__global__ __launch_bounds__(256) void cvtT_kernel(
    const float* __restrict__ w, _Float16* __restrict__ wT, int K, int N, int ldt)
{
  long idx = (long)blockIdx.x * blockDim.x + threadIdx.x;
  if (idx >= (long)K * N) return;
  int n = (int)(idx % N);
  int k = (int)(idx / N);
  wT[(size_t)n * ldt + k] = (_Float16)w[idx];
}

// dst[(b*dst_bstride + r)*cols + c] = src[r*cols + c]
__global__ __launch_bounds__(256) void bcast_kernel(
    const float* __restrict__ src, float* __restrict__ dst,
    int rows, int cols, long dst_bstride, int nb)
{
  int idx = blockIdx.x * blockDim.x + threadIdx.x;
  int total = nb * rows * cols;
  if (idx >= total) return;
  int c = idx % cols;
  int t = idx / cols;
  int r = t % rows;
  int b = t / rows;
  dst[((long)b * dst_bstride + r) * cols + c] = src[(long)r * cols + c];
}

// ---------------------------------------------------------------------------
extern "C" void kernel_launch(void* const* d_in, const int* in_sizes, int n_in,
                              void* d_out, int out_size, void* d_ws, size_t ws_size,
                              hipStream_t stream)
{
  const float* rna        = (const float*)d_in[0];
  const float* atac       = (const float*)d_in[1];
  const float* rna_ln1_g  = (const float*)d_in[2];
  const float* rna_ln1_b  = (const float*)d_in[3];
  const float* rna_w      = (const float*)d_in[4];
  const float* rna_b      = (const float*)d_in[5];
  const float* rna_ln2_g  = (const float*)d_in[6];
  const float* rna_ln2_b  = (const float*)d_in[7];
  const float* atac_ln1_g = (const float*)d_in[8];
  const float* atac_ln1_b = (const float*)d_in[9];
  const float* atac_w     = (const float*)d_in[10];
  const float* atac_b     = (const float*)d_in[11];
  const float* atac_ln2_g = (const float*)d_in[12];
  const float* atac_ln2_b = (const float*)d_in[13];
  const float* fusion_tok = (const float*)d_in[14];
  const float* lay_ag     = (const float*)d_in[15];
  const float* lay_wq     = (const float*)d_in[16];
  const float* lay_wkv    = (const float*)d_in[17];
  const float* lay_wo     = (const float*)d_in[18];
  const float* lay_fg     = (const float*)d_in[19];
  const float* lay_w1     = (const float*)d_in[20];
  const float* lay_w2     = (const float*)d_in[21];
  const float* final_g    = (const float*)d_in[22];
  const float* ret_tok    = (const float*)d_in[23];
  const float* pool_g     = (const float*)d_in[24];
  const float* pool_wq    = (const float*)d_in[25];
  const float* pool_wkv   = (const float*)d_in[26];
  const float* pool_wo    = (const float*)d_in[27];

  const int MT = ZB * ZNTOK;                 // 3104 token rows
  char* base = (char*)d_ws;
  size_t off = 0;
  auto alloc = [&](size_t bytes) { void* p = base + off; off = (off + bytes + 255) & ~(size_t)255; return p; };

  float*    tok     = (float*)alloc((size_t)MT * ZDIM * 4);
  float*    bufC    = (float*)alloc((size_t)MT * ZDIM * 4);           // embed proj (f32, LN input)
  float*    bufFF   = (float*)alloc((size_t)MT * 2 * ZIFF * 4);
  float*    retb    = (float*)alloc((size_t)6 * ZDIM * 4);
  _Float16* bufA_h  = (_Float16*)alloc((size_t)MT * ZDIM * 2);        // LN f16 outs (also 1536x1024 embed LN)
  _Float16* bufQ_h  = (_Float16*)alloc((size_t)MT * ZDIM * 2);
  _Float16* bufKV_h = (_Float16*)alloc((size_t)MT * 2 * ZDIM * 2);
  _Float16* bufAtt_h= (_Float16*)alloc((size_t)MT * ZDIM * 2);
  _Float16* bufG_h  = (_Float16*)alloc((size_t)MT * ZIFFP * 2);
  _Float16* retn_h  = (_Float16*)alloc((size_t)3 * ZDIM * 2);
  _Float16* retq_h  = (_Float16*)alloc((size_t)3 * ZDIM * 2);
  // f16 transposed weights [N,K]
  _Float16* rna_wT  = (_Float16*)alloc((size_t)ZDIM * ZRIN * 2);
  _Float16* atac_wT = (_Float16*)alloc((size_t)ZDIM * ZRIN * 2);
  _Float16 *wqT[ZDEPTH], *wkvT[ZDEPTH], *woT[ZDEPTH], *w1T[ZDEPTH], *w2T[ZDEPTH];
  for (int l = 0; l < ZDEPTH; ++l) {
    wqT[l]  = (_Float16*)alloc((size_t)ZDIM * ZDIM * 2);
    wkvT[l] = (_Float16*)alloc((size_t)2 * ZDIM * ZDIM * 2);
    woT[l]  = (_Float16*)alloc((size_t)ZDIM * ZDIM * 2);
    w1T[l]  = (_Float16*)alloc((size_t)2 * ZIFF * ZDIM * 2);
    w2T[l]  = (_Float16*)alloc((size_t)ZDIM * ZIFFP * 2);
  }
  _Float16* pool_wqT  = (_Float16*)alloc((size_t)ZDIM * ZDIM * 2);
  _Float16* pool_wkvT = (_Float16*)alloc((size_t)2 * ZDIM * ZDIM * 2);
  _Float16* pool_woT  = (_Float16*)alloc((size_t)ZDIM * ZDIM * 2);

  auto cvtT = [&](const float* w, _Float16* wT, int K, int N, int ldt) {
    long total = (long)K * N;
    cvtT_kernel<<<(int)((total + 255) / 256), 256, 0, stream>>>(w, wT, K, N, ldt);
  };
  auto gemm = [&](const _Float16* A, long lda, const _Float16* Bm, long ldb,
                  const float* bias, const float* resid, float* C, _Float16* Ch,
                  int M, int N, int K, float alpha) {
    dim3 g((N + GBN - 1) / GBN, (M + GBM - 1) / GBM);
    gemm_kernel<<<g, 256, 0, stream>>>(A, lda, Bm, ldb, bias, resid, C, Ch, M, N, K, alpha);
  };
  auto ln = [&](const float* in, float* o32, _Float16* o16, const float* g, const float* b,
                int rows, int cols, int rpb, long obs, int oro) {
    ln_kernel<<<rows, 256, 0, stream>>>(in, o32, o16, g, b, rows, cols, rpb, obs, oro);
  };

  // ---- weight conversion (f32 -> f16, transposed to [N,K]) ----
  cvtT(rna_w,  rna_wT,  ZRIN, ZDIM, ZRIN);
  cvtT(atac_w, atac_wT, ZRIN, ZDIM, ZRIN);
  for (int l = 0; l < ZDEPTH; ++l) {
    cvtT(lay_wq  + (size_t)l * ZDIM * ZDIM,     wqT[l],  ZDIM, ZDIM,     ZDIM);
    cvtT(lay_wkv + (size_t)l * ZDIM * 2 * ZDIM, wkvT[l], ZDIM, 2 * ZDIM, ZDIM);
    cvtT(lay_wo  + (size_t)l * ZDIM * ZDIM,     woT[l],  ZDIM, ZDIM,     ZDIM);
    cvtT(lay_w1  + (size_t)l * ZDIM * 2 * ZIFF, w1T[l],  ZDIM, 2 * ZIFF, ZDIM);
    cvtT(lay_w2  + (size_t)l * ZIFF * ZDIM,     w2T[l],  ZIFF, ZDIM,     ZIFFP);
  }
  cvtT(pool_wq,  pool_wqT,  ZDIM, ZDIM,     ZDIM);
  cvtT(pool_wkv, pool_wkvT, ZDIM, 2 * ZDIM, ZDIM);
  cvtT(pool_wo,  pool_woT,  ZDIM, ZDIM,     ZDIM);

  // ---- embeds: LN1(f16) -> GEMM+bias(f32) -> LN2 scattered into tok ----
  ln(rna, nullptr, bufA_h, rna_ln1_g, rna_ln1_b, ZB * ZNR, ZRIN, ZB * ZNR, 0, 0);
  gemm(bufA_h, ZRIN, rna_wT, ZRIN, rna_b, nullptr, bufC, nullptr, ZB * ZNR, ZDIM, ZRIN, 1.f);
  ln(bufC, tok, nullptr, rna_ln2_g, rna_ln2_b, ZB * ZNR, ZDIM, ZNR, ZNTOK, 0);

  ln(atac, nullptr, bufA_h, atac_ln1_g, atac_ln1_b, ZB * ZNA, ZRIN, ZB * ZNA, 0, 0);
  gemm(bufA_h, ZRIN, atac_wT, ZRIN, atac_b, nullptr, bufC, nullptr, ZB * ZNA, ZDIM, ZRIN, 1.f);
  ln(bufC, tok, nullptr, atac_ln2_g, atac_ln2_b, ZB * ZNA, ZDIM, ZNA, ZNTOK, ZNR + ZNF);

  {
    int total = ZB * ZNF * ZDIM;
    bcast_kernel<<<(total + 255) / 256, 256, 0, stream>>>(
        fusion_tok, tok + (size_t)ZNR * ZDIM, ZNF, ZDIM, ZNTOK, ZB);
  }

  // ---- transformer layers ----
  const float qscale = 0.125f;  // DHEAD^-0.5
  for (int l = 0; l < ZDEPTH; ++l) {
    const float* ag = lay_ag + (size_t)l * ZDIM;
    const float* fg = lay_fg + (size_t)l * ZDIM;

    ln(tok, nullptr, bufA_h, ag, nullptr, MT, ZDIM, MT, 0, 0);                            // xn (f16)
    gemm(bufA_h, ZDIM, wqT[l],  ZDIM, nullptr, nullptr, nullptr, bufQ_h,  MT, ZDIM,     ZDIM, qscale);
    gemm(bufA_h, ZDIM, wkvT[l], ZDIM, nullptr, nullptr, nullptr, bufKV_h, MT, 2 * ZDIM, ZDIM, 1.f);
    attn_kernel<<<dim3(ZKB, ZH, ZB), 256, 0, stream>>>(
        bufQ_h, (long)ZNTOK * ZDIM, ZNTOK, bufKV_h, (long)ZNTOK * 2 * ZDIM,
        bufAtt_h, (long)ZNTOK * ZDIM, /*mask_mode=*/0);
    gemm(bufAtt_h, ZDIM, woT[l], ZDIM, nullptr, tok, tok, nullptr, MT, ZDIM, ZDIM, 1.f); // +resid

    ln(tok, nullptr, bufA_h, fg, nullptr, MT, ZDIM, MT, 0, 0);                            // h (f16)
    gemm(bufA_h, ZDIM, w1T[l], ZDIM, nullptr, nullptr, bufFF, nullptr, MT, 2 * ZIFF, ZDIM, 1.f);
    {
      long total = (long)MT * ZIFF;
      geglu_kernel<<<(int)((total + 255) / 256), 256, 0, stream>>>(bufFF, bufG_h, MT);
    }
    gemm(bufG_h, ZIFFP, w2T[l], ZIFFP, nullptr, tok, tok, nullptr, MT, ZDIM, ZIFF, 1.f); // +resid
  }

  // ---- final LN + attention pool ----
  ln(tok, nullptr, bufA_h, final_g, nullptr, MT, ZDIM, MT, 0, 0);                         // tokens_n (f16)

  ln(ret_tok, nullptr, retn_h, pool_g, nullptr, 3, ZDIM, 3, 0, 0);
  gemm(retn_h, ZDIM, pool_wqT,  ZDIM, nullptr, nullptr, nullptr, retq_h,  3,  ZDIM,     ZDIM, qscale);
  gemm(bufA_h, ZDIM, pool_wkvT, ZDIM, nullptr, nullptr, nullptr, bufKV_h, MT, 2 * ZDIM, ZDIM, 1.f);
  attn_kernel<<<dim3(1, ZH, ZB), 256, 0, stream>>>(
      retq_h, 0L, 3, bufKV_h, (long)ZNTOK * 2 * ZDIM,
      bufAtt_h, (long)3 * ZDIM, /*mask_mode=*/1);

  bcast_kernel<<<(ZB * 3 * ZDIM + 255) / 256, 256, 0, stream>>>(ret_tok, retb, 3, ZDIM, 3, ZB);
  gemm(bufAtt_h, ZDIM, pool_woT, ZDIM, nullptr, retb, (float*)d_out, nullptr, ZB * 3, ZDIM, ZDIM, 1.f);
}
